// ThirdOrderAttention_4741643894808
// MI455X (gfx1250) — matrix operand order repaired and checked by reading, not compiled, not run on hardware
//
#include <hip/hip_runtime.h>
#include <hip/hip_fp16.h>

typedef __attribute__((ext_vector_type(16))) _Float16 v16h;
typedef __attribute__((ext_vector_type(8)))  _Float16 h8;
typedef __attribute__((ext_vector_type(8)))  float    v8f;

#define BB 2
#define TT 256
#define CC 512
#define HH 8
#define DD 64
#define BH (BB*HH)   // 16

__device__ __forceinline__ v16h cat8(h8 lo, h8 hi) {
    return __builtin_shufflevector(lo, hi, 0, 1, 2, 3, 4, 5, 6, 7,
                                   8, 9, 10, 11, 12, 13, 14, 15);
}

// ---------------------------------------------------------------------------
// Kernel 1: five input projections y = x @ W, stored head-major f16.
// Block = 128 (4 waves); all 4 waves share the same 16 A rows -> stage them
// once in LDS (coalesced), then vectorized b128 fragment loads.
// ---------------------------------------------------------------------------
__global__ __launch_bounds__(128)
void proj_kernel(const float* __restrict__ x,
                 const float* __restrict__ Wq,  const float* __restrict__ Wk0,
                 const float* __restrict__ Wk1, const float* __restrict__ Wv0,
                 const float* __restrict__ Wv1,
                 _Float16* __restrict__ proj)
{
    __shared__ __align__(16) _Float16 x_s[16 * 520];   // padded stride 520

    const int which = blockIdx.y;
    const float* W = (which == 0) ? Wq : (which == 1) ? Wk0 :
                     (which == 2) ? Wk1 : (which == 3) ? Wv0 : Wv1;
    const int tid  = threadIdx.x;
    const int wave = tid >> 5;
    const int lane = tid & 31;
    const int lane16 = lane & 15, grp = lane >> 4;
    const int tile0 = blockIdx.x * 4;
    const int mt = tile0 >> 5;                 // same for all 4 waves
    const int nt = (tile0 + wave) & 31;
    const int n  = nt * 16 + lane16;

    for (int i = tid; i < 16 * 512; i += 128)
        x_s[(i >> 9) * 520 + (i & 511)] =
            (_Float16)x[(mt * 16 + (i >> 9)) * CC + (i & 511)];
    __syncthreads();

    const h8* xv = (const h8*)(x_s + lane16 * 520);    // 520/8 = 65 h8/row
    const float* Wn = W + (size_t)(grp * 16) * CC + n;

    v8f c;
    #pragma unroll
    for (int e = 0; e < 8; ++e) c[e] = 0.f;

    #pragma unroll 1
    for (int kc = 0; kc < 16; ++kc) {
        v16h a = cat8(xv[kc * 4 + grp * 2], xv[kc * 4 + grp * 2 + 1]);
        v16h b;
        #pragma unroll
        for (int j = 0; j < 16; ++j)
            b[j] = (_Float16)Wn[(size_t)(kc * 32 + j) * CC];
        c = __builtin_amdgcn_wmma_f32_16x16x32_f16(false, a, false, b,
                                                   (short)0, c, false, false);
    }

    // Store: head-major scatter collapses to base + r*DD immediate offsets
    // (the 8 rows mm0..mm0+7 never cross the batch boundary).
    {
        const int mm0 = mt * 16 + grp * 8;
        const int b_ = mm0 >> 8, t0 = mm0 & 255, h = n >> 6, d = n & 63;
        _Float16* pbase = proj + (size_t)which * (BH * TT * DD) +
                          (((size_t)(b_ * HH + h) * TT + t0) * DD + d);
        #pragma unroll
        for (int r = 0; r < 8; ++r)
            pbase[r * DD] = (_Float16)c[r];
    }
}

// ---------------------------------------------------------------------------
// Kernel 2: trilinear attention core.
// grid = (bh=16, l-chunk=16); block = 512 (16 waves); wave w owns k-tile w.
// Per l: S = (q .* k0[l]) @ k1^T, softmax over m, p_attn store,
//        U = P @ v1, out_acc += U .* v0[l].  out_acc atomically added at end.
// ---------------------------------------------------------------------------
__global__ __launch_bounds__(512)
void attn_kernel(const _Float16* __restrict__ proj,
                 float* __restrict__ attn_acc,
                 float* __restrict__ p_out)
{
    __shared__ __align__(16) _Float16 q_s  [TT * 72];      // 36 KB, stride 72
    __shared__ __align__(16) _Float16 k1_s [TT * 72];      // 36 KB, stride 72
    __shared__ __align__(16) _Float16 v1t_s[DD * 264];     // 33 KB, transposed
    __shared__ __align__(16) _Float16 p_s  [16 * 16 * 40]; // 20 KB, stride 40
    __shared__ __align__(16) _Float16 k0l_s[DD];
    __shared__ __align__(16) float    v0l_s[DD];

    const int bh   = blockIdx.x;
    const int lch  = blockIdx.y;
    const int tid  = threadIdx.x;
    const int wave = tid >> 5;
    const int lane = tid & 31;
    const int lane16 = lane & 15, grp = lane >> 4;
    const int kt = wave;

    const size_t hs = (size_t)BH * TT * DD;
    const _Float16* qg  = proj + 0 * hs + (size_t)bh * TT * DD;
    const _Float16* k0g = proj + 1 * hs + (size_t)bh * TT * DD;
    const _Float16* k1g = proj + 2 * hs + (size_t)bh * TT * DD;
    const _Float16* v0g = proj + 3 * hs + (size_t)bh * TT * DD;
    const _Float16* v1g = proj + 4 * hs + (size_t)bh * TT * DD;

    for (int i = tid; i < TT * DD; i += 512) {
        int t = i >> 6, d = i & 63;
        q_s  [t * 72 + d]  = qg[i];
        k1_s [t * 72 + d]  = k1g[i];
        v1t_s[d * 264 + t] = v1g[i];
    }
    __syncthreads();

    v8f out_acc[4];
    #pragma unroll
    for (int dt = 0; dt < 4; ++dt)
        #pragma unroll
        for (int e = 0; e < 8; ++e) out_acc[dt][e] = 0.f;

    _Float16* pw = p_s + wave * (16 * 40);
    const h8* qv  = (const h8*)(q_s + (kt * 16 + lane16) * 72);  // 9 h8/row
    const h8* k0v = (const h8*)k0l_s;
    const h8* pv  = (const h8*)(pw + lane16 * 40);               // 5 h8/row

    #pragma unroll 1
    for (int il = 0; il < 16; ++il) {
        const int l = lch * 16 + il;
        __syncthreads();
        if (tid < DD) {
            k0l_s[tid] = k0g[l * DD + tid];
            v0l_s[tid] = (float)v0g[l * DD + tid];
        }
        if (il < 15 && tid == 0) {
            __builtin_prefetch(&k0g[(l + 1) * DD], 0, 3);
            __builtin_prefetch(&v0g[(l + 1) * DD], 0, 3);
        }
        __syncthreads();

        // A fragments: A[kr][d] = q[kt*16+kr][d] * k0[l][d]  (packed f16 mul)
        v16h afrag[2];
        #pragma unroll
        for (int kc = 0; kc < 2; ++kc) {
            h8 lo = qv[kc * 4 + grp]     * k0v[kc * 4 + grp];
            h8 hi = qv[kc * 4 + grp + 2] * k0v[kc * 4 + grp + 2];
            afrag[kc] = cat8(lo, hi);
        }

        // S = A @ k1^T : 16 m-tiles, K=64 in two chunks of 32
        v8f c[16];
        #pragma unroll
        for (int mt = 0; mt < 16; ++mt) {
            v8f acc;
            #pragma unroll
            for (int e = 0; e < 8; ++e) acc[e] = 0.f;
            const h8* kr = (const h8*)(k1_s + (mt * 16 + lane16) * 72);
            #pragma unroll
            for (int kc = 0; kc < 2; ++kc) {
                v16h b = cat8(kr[kc * 4 + grp], kr[kc * 4 + grp + 2]);
                acc = __builtin_amdgcn_wmma_f32_16x16x32_f16(false, afrag[kc],
                        false, b, (short)0, acc, false, false);
            }
            c[mt] = acc;
        }

        // softmax over m (scale, row-max/sum across 16 tiles + 16 lanes)
        float rowmax[8], inv[8];
        #pragma unroll
        for (int r = 0; r < 8; ++r) {
            float mx = -1e30f;
            #pragma unroll
            for (int mt = 0; mt < 16; ++mt) {
                float v = c[mt][r] * 0.125f;   // 1/sqrt(64)
                c[mt][r] = v;
                mx = fmaxf(mx, v);
            }
            #pragma unroll
            for (int s = 1; s < 16; s <<= 1) mx = fmaxf(mx, __shfl_xor(mx, s));
            rowmax[r] = mx;
        }
        #pragma unroll
        for (int r = 0; r < 8; ++r) {
            float sum = 0.f;
            #pragma unroll
            for (int mt = 0; mt < 16; ++mt) {
                float e = __expf(c[mt][r] - rowmax[r]);
                c[mt][r] = e;
                sum += e;
            }
            #pragma unroll
            for (int s = 1; s < 16; s <<= 1) sum += __shfl_xor(sum, s);
            inv[r] = 1.0f / sum;
        }

        // p_attn base for this lane/l; stores use constant element offsets
        float* pb = p_out + (((size_t)bh * TT + kt * 16 + grp * 8) * TT + l) * TT
                  + lane16;

        // U = P @ v1 in chunks of 2 m-tiles (K=32 per WMMA)
        v8f u[4];
        #pragma unroll
        for (int dt = 0; dt < 4; ++dt)
            #pragma unroll
            for (int e = 0; e < 8; ++e) u[dt][e] = 0.f;

        #pragma unroll
        for (int ch = 0; ch < 8; ++ch) {
            #pragma unroll
            for (int t2 = 0; t2 < 2; ++t2) {
                const int mt = ch * 2 + t2;
                #pragma unroll
                for (int r = 0; r < 8; ++r) {
                    float p = c[mt][r] * inv[r];
                    pb[r * TT * TT + mt * 16] = p;               // p_attn (f32)
                    pw[(grp * 8 + r) * 40 + t2 * 16 + lane16] = (_Float16)p;
                }
            }
            asm volatile("s_wait_dscnt 0" ::: "memory");
            v16h a2 = cat8(pv[grp], pv[grp + 2]);
            #pragma unroll
            for (int dt = 0; dt < 4; ++dt) {
                const h8* vr = (const h8*)(v1t_s + (dt * 16 + lane16) * 264);
                v16h b2 = cat8(vr[ch * 4 + grp], vr[ch * 4 + grp + 2]);
                u[dt] = __builtin_amdgcn_wmma_f32_16x16x32_f16(false, a2,
                          false, b2, (short)0, u[dt], false, false);
            }
        }

        // out_acc += U .* v0[l]
        #pragma unroll
        for (int dt = 0; dt < 4; ++dt) {
            float v0v = v0l_s[dt * 16 + lane16];
            #pragma unroll
            for (int r = 0; r < 8; ++r)
                out_acc[dt][r] += u[dt][r] * v0v;
        }
    }

    {
        float* abase = attn_acc +
            ((size_t)bh * TT + kt * 16 + grp * 8) * DD + lane16;
        #pragma unroll
        for (int dt = 0; dt < 4; ++dt)
            #pragma unroll
            for (int r = 0; r < 8; ++r)
                atomicAdd(&abase[r * DD + dt * 16], out_acc[dt][r]);
    }
}

// ---------------------------------------------------------------------------
// Kernel 3: out = attn @ W_o + b_o (attn re-gathered head-major -> LDS tile)
// ---------------------------------------------------------------------------
__global__ __launch_bounds__(128)
void outproj_kernel(const float* __restrict__ attn_acc,
                    const float* __restrict__ Wo,
                    const float* __restrict__ bo,
                    float* __restrict__ out)
{
    __shared__ __align__(16) _Float16 a_s[16 * 520];

    const int tid  = threadIdx.x;
    const int wave = tid >> 5;
    const int lane = tid & 31;
    const int lane16 = lane & 15, grp = lane >> 4;
    const int tile0 = blockIdx.x * 4;
    const int mt = tile0 >> 5;
    const int nt = (tile0 + wave) & 31;
    const int n  = nt * 16 + lane16;

    for (int i = tid; i < 16 * 512; i += 128) {
        int row = i >> 9, kk = i & 511;
        int mm = mt * 16 + row;
        int b_ = mm >> 8, t = mm & 255, h = kk >> 6, d = kk & 63;
        a_s[row * 520 + kk] =
            (_Float16)attn_acc[(((size_t)(b_ * HH + h) * TT + t) * DD + d)];
    }
    __syncthreads();

    const h8* av = (const h8*)(a_s + lane16 * 520);
    const float* Wn = Wo + (size_t)(grp * 16) * CC + n;

    v8f c;
    #pragma unroll
    for (int e = 0; e < 8; ++e) c[e] = 0.f;

    #pragma unroll 1
    for (int kc = 0; kc < 16; ++kc) {
        v16h a = cat8(av[kc * 4 + grp * 2], av[kc * 4 + grp * 2 + 1]);
        v16h b;
        #pragma unroll
        for (int j = 0; j < 16; ++j)
            b[j] = (_Float16)Wn[(size_t)(kc * 32 + j) * CC];
        c = __builtin_amdgcn_wmma_f32_16x16x32_f16(false, a, false, b,
                                                   (short)0, c, false, false);
    }

    float bias = bo[n];
    float* obase = out + (size_t)(mt * 16 + grp * 8) * CC + n;
    #pragma unroll
    for (int r = 0; r < 8; ++r)
        obase[r * CC] = c[r] + bias;
}

// ---------------------------------------------------------------------------
extern "C" void kernel_launch(void* const* d_in, const int* in_sizes, int n_in,
                              void* d_out, int out_size, void* d_ws, size_t ws_size,
                              hipStream_t stream)
{
    (void)in_sizes; (void)n_in; (void)out_size; (void)ws_size;

    const float* x   = (const float*)d_in[0];
    const float* Wq  = (const float*)d_in[1];
    const float* Wk0 = (const float*)d_in[2];
    const float* Wk1 = (const float*)d_in[3];
    const float* Wv0 = (const float*)d_in[4];
    const float* Wv1 = (const float*)d_in[5];
    const float* Wo  = (const float*)d_in[6];
    const float* bo  = (const float*)d_in[7];

    float* out   = (float*)d_out;                  // (B,T,C) f32
    float* p_out = out + (size_t)BB * TT * CC;     // (B,H,T,T,T) f32

    _Float16* proj = (_Float16*)d_ws;              // 5 * 262144 halves
    float* attn_acc = (float*)((char*)d_ws +
                      5 * (size_t)BH * TT * DD * sizeof(_Float16));

    hipMemsetAsync(attn_acc, 0, (size_t)BH * TT * DD * sizeof(float), stream);

    proj_kernel<<<dim3(256, 5), 128, 0, stream>>>(x, Wq, Wk0, Wk1, Wv0, Wv1, proj);
    attn_kernel<<<dim3(16, 16), 512, 0, stream>>>(proj, attn_acc, p_out);
    outproj_kernel<<<256, 128, 0, stream>>>(attn_acc, Wo, bo, out);
}
